// HahnPolynomials_22591527977628
// MI455X (gfx1250) — matrix operand order repaired and checked by reading, not compiled, not run on hardware
//
#include <hip/hip_runtime.h>

// ---------------------------------------------------------------------------
// Hahn polynomial layer as a fused tanh + recurrence + WMMA GEMM for gfx1250.
//   y[n,o] = sum_{i,d} T_d(tanh(x[n,i])) * coeffs[i,o,d]
// GEMM view: M=32768, N=512, K=512*9=4608, f16 inputs, f32 accumulate.
// Pipelined: double-buffered LDS, async global->LDS B copies (ASYNCcnt),
// one barrier per K-step, 2x8 output tiles per wave with B-fragment reuse.
// Recurrence state ping-pongs between two register arrays (no tc temp).
// ---------------------------------------------------------------------------

typedef __attribute__((ext_vector_type(16))) _Float16 v16h;
typedef __attribute__((ext_vector_type(8)))  _Float16 v8h;
typedef __attribute__((ext_vector_type(8)))  float    v8f;
typedef int vint4 __attribute__((vector_size(16)));   // async builtin pointee

#define INPUT_DIM  512
#define OUTPUT_DIM 512
#define DEGREE     8
#define NDEG       (DEGREE + 1)     // 9
#define KI         32               // K per WMMA step (i-block width)
#define NIB        (INPUT_DIM / KI) // 16 i-blocks
#define ROWS       128              // rows (M) per workgroup
#define COLS       256              // output cols (N) per workgroup
#define THREADS    256              // 8 waves, arranged 4 (M) x 2 (N)
#define RPT        16               // basis values per thread
#define LDA        40               // padded halves per ldsA row (80B, 16B aligned)
#define LDB        40               // padded halves per ldsB row
#define A_TILE     (ROWS * LDA)     // 5120 halves = 10240 B
#define B_TILE     (COLS * LDB)     // 10240 halves = 20480 B

#define ALPHA_C 0.5f
#define BETA_C  0.5f
#define NPAR_C  32.0f

#if defined(__clang__) && __has_builtin(__builtin_amdgcn_tanhf)
__device__ inline float fast_tanh(float v) { return __builtin_amdgcn_tanhf(v); }
#else
__device__ inline float fast_tanh(float v) { return tanhf(v); }
#endif

__device__ inline v16h cat16(v8h lo, v8h hi) {
  return __builtin_shufflevector(lo, hi, 0, 1, 2, 3, 4, 5, 6, 7,
                                         8, 9, 10, 11, 12, 13, 14, 15);
}

// --- CDNA5 async global->LDS copy (ASYNCcnt-tracked), 16B per lane ---------
__device__ __forceinline__ void async_copy_b128(const _Float16* g, _Float16* l) {
#if __has_builtin(__builtin_amdgcn_global_load_async_to_lds_b128)
  // Prototype (from hipcc diagnostic): (int4* src, int4* dst, imm offset, imm cpol)
  __builtin_amdgcn_global_load_async_to_lds_b128((vint4*)g, (vint4*)l, 0, 0);
#else
  asm volatile("global_load_async_to_lds_b128 %0, %1, off"
               :: "v"((unsigned)(__SIZE_TYPE__)l), "v"(g) : "memory");
#endif
}

__device__ __forceinline__ void wait_async0() {
#if __has_builtin(__builtin_amdgcn_s_wait_asynccnt)
  __builtin_amdgcn_s_wait_asynccnt(0);
#else
  asm volatile("s_wait_asynccnt 0x0" ::: "memory");
#endif
}

// Three-term Hahn recurrence step: dst currently holds T_{dd-2}, src holds
// T_{dd-1}; overwrite dst with T_dd.  DDV is compile-time -> constants fold.
#define RECUR(dst, src, DDV)                                                   \
  do {                                                                         \
    const float m_  = (float)((DDV) - 1);                                      \
    const float Ax_ = (m_ + ALPHA_C + BETA_C + 1.f) * (m_ + ALPHA_C + 1.f) *   \
                      (NPAR_C - m_) /                                          \
                      ((2.f * m_ + ALPHA_C + BETA_C + 1.f) *                   \
                       (2.f * m_ + ALPHA_C + BETA_C + 2.f));                   \
    const float Cx_ = m_ * (m_ + ALPHA_C + BETA_C + NPAR_C + 1.f) *            \
                      (m_ + BETA_C) /                                          \
                      ((2.f * m_ + ALPHA_C + BETA_C) *                         \
                       (2.f * m_ + ALPHA_C + BETA_C + 1.f));                   \
    const float rAx_ = 1.f / Ax_;                                              \
    _Pragma("unroll")                                                          \
    for (int rep = 0; rep < RPT; ++rep)                                        \
      dst[rep] = ((Ax_ + Cx_ - x2v[rep]) * src[rep] - Cx_ * dst[rep]) * rAx_;  \
  } while (0)

// ---------------------------------------------------------------------------
// Kernel 1: pack coeffs[i,o,d] (f32, stride-9 in d) into f16 workspace laid
// out [d][ib][o][kk] so the GEMM kernel's B-tile fills are coalesced 64B rows.
// ---------------------------------------------------------------------------
__global__ void hahn_pack_coeffs(const float* __restrict__ coeffs,
                                 _Float16* __restrict__ bws) {
  int idx = blockIdx.x * blockDim.x + threadIdx.x;
  const int total = NDEG * NIB * OUTPUT_DIM * KI;
  if (idx >= total) return;
  int kk = idx & (KI - 1);
  int o  = (idx / KI) & (OUTPUT_DIM - 1);
  int ib = (idx / (KI * OUTPUT_DIM)) & (NIB - 1);
  int d  =  idx / (KI * OUTPUT_DIM * NIB);
  int i  = ib * KI + kk;
  bws[idx] = (_Float16)coeffs[(i * OUTPUT_DIM + o) * NDEG + d];
}

// ---------------------------------------------------------------------------
// Kernel 2: fused tanh + Hahn recurrence + pipelined WMMA GEMM.
// ---------------------------------------------------------------------------
__launch_bounds__(THREADS)
__global__ void hahn_wmma(const float* __restrict__ x,
                          const _Float16* __restrict__ bws,
                          float* __restrict__ out) {
  __shared__ __align__(16) _Float16 ldsA[2][A_TILE];  // 20 KB
  __shared__ __align__(16) _Float16 ldsB[2][B_TILE];  // 40 KB

  const int tid   = threadIdx.x;
  const int lane  = tid & 31;
  const int wv    = tid >> 5;      // 0..7
  const int mrow  = wv >> 1;       // 0..3: owns M-tiles {2*mrow, 2*mrow+1}
  const int ncol  = wv & 1;        // 0..1: owns 128-col half
  const int l16   = lane & 15;
  const int lhalf = lane >> 4;
  const int rowBase = blockIdx.x * ROWS;
  const int oBase   = blockIdx.y * COLS;

  const float C1 = (ALPHA_C + BETA_C + 2.f) / ((ALPHA_C + 1.f) * NPAR_C);

  v8f acc[16];
#pragma unroll
  for (int t = 0; t < 16; ++t) {
    v8f z = {0.f, 0.f, 0.f, 0.f, 0.f, 0.f, 0.f, 0.f};
    acc[t] = z;
  }

  // Recurrence state: thread owns rows r = wv + 8*rep (rep=0..15), col i0+lane.
  // tA holds T_d for even d, tB for odd d (ping-pong; no temp array).
  float x2v[RPT], tA[RPT], tB[RPT];
#pragma unroll
  for (int rep = 0; rep < RPT; ++rep) { tA[rep] = 0.f; tB[rep] = 0.f; }

  // Prepare K-step (degree dd, i-block ibn) into LDS buffer bufn.
  auto prepare = [&](int dd, int ibn, int bufn) {
    if (dd == 0) {
      const float* xp = x + (size_t)(rowBase + wv) * INPUT_DIM + ibn * KI + lane;
#pragma unroll
      for (int rep = 0; rep < RPT; ++rep)
        x2v[rep] = fast_tanh(xp[rep * 8 * INPUT_DIM]);
      if (ibn + 1 < NIB) {  // pull next i-block toward L2/L0 ~9 K-steps early
#pragma unroll
        for (int rep = 0; rep < RPT; ++rep)
          __builtin_prefetch(xp + KI + rep * 8 * INPUT_DIM);
      }
    }
    // T_dd into tA (even dd) or tB (odd dd); all dd-branches compile-time.
    if (dd == 0) {
#pragma unroll
      for (int rep = 0; rep < RPT; ++rep) tA[rep] = 1.f;
    } else if (dd == 1) {
#pragma unroll
      for (int rep = 0; rep < RPT; ++rep) tB[rep] = 1.f - C1 * x2v[rep];
    } else if ((dd & 1) == 0) {
      RECUR(tA, tB, dd);
    } else {
      RECUR(tB, tA, dd);
    }
    // A tile: row-major halves, matches 16-bit A 16x32 fragment layout.
    _Float16* sA = &ldsA[bufn][0];
    if ((dd & 1) == 0) {
#pragma unroll
      for (int rep = 0; rep < RPT; ++rep)
        sA[(wv + 8 * rep) * LDA + lane] = (_Float16)tA[rep];
    } else {
#pragma unroll
      for (int rep = 0; rep < RPT; ++rep)
        sA[(wv + 8 * rep) * LDA + lane] = (_Float16)tB[rep];
    }
    // B tile: thread tid owns o = oBase+tid, 64B via 4 async b128 copies.
    const _Float16* gsrc =
        bws + (size_t)((dd * NIB + ibn) * OUTPUT_DIM + oBase + tid) * KI;
    _Float16* ldst = &ldsB[bufn][tid * LDB];
#pragma unroll
    for (int q = 0; q < 4; ++q)
      async_copy_b128(gsrc + q * 8, ldst + q * 8);
  };

  prepare(0, 0, 0);

  for (int ib = 0; ib < NIB; ++ib) {
#pragma unroll
    for (int d = 0; d < NDEG; ++d) {
      const int buf = (ib + d) & 1;     // s = ib*9 + d; parity = (ib+d)&1
      wait_async0();                     // B(buf) copies done (own wave)
      __syncthreads();                   // all waves' A/B(buf) visible

      // Kick off next step's recurrence + A store + async B copy: overlaps
      // with this step's WMMAs.
      if (d + 1 < NDEG)      prepare(d + 1, ib, buf ^ 1);
      else if (ib + 1 < NIB) prepare(0, ib + 1, buf ^ 1);

      // A fragments: lanes 0-15 row=l16 K0-15; lanes 16-31 K16-31.
      const _Float16* sA = &ldsA[buf][0];
      const _Float16* ap0 = sA + (mrow * 32 + l16) * LDA + lhalf * 16;
      v16h a0 = cat16(*(const v8h*)(ap0), *(const v8h*)(ap0 + 8));
      const _Float16* ap1 = ap0 + 16 * LDA;
      v16h a1 = cat16(*(const v8h*)(ap1), *(const v8h*)(ap1 + 8));

      // B fragments: col=l16, K = lhalf*16 + j; one-ahead register pipeline,
      // each fragment feeds two WMMAs (both M-tiles).
      const _Float16* bbase = &ldsB[buf][0] + (ncol * 128 + l16) * LDB + lhalf * 16;
      v16h bc = cat16(*(const v8h*)(bbase), *(const v8h*)(bbase + 8));
#pragma unroll
      for (int nt = 0; nt < 8; ++nt) {
        v16h bn = bc;
        if (nt + 1 < 8) {
          const _Float16* bp = bbase + (nt + 1) * (16 * LDB);
          bn = cat16(*(const v8h*)(bp), *(const v8h*)(bp + 8));
        }
        acc[nt] = __builtin_amdgcn_wmma_f32_16x16x32_f16(
            false, a0, false, bc, (short)0, acc[nt], false, false);
        acc[8 + nt] = __builtin_amdgcn_wmma_f32_16x16x32_f16(
            false, a1, false, bc, (short)0, acc[8 + nt], false, false);
        bc = bn;
      }
    }
  }

  // Store: C/D f32 layout, VGPR j -> row j + 8*lhalf, col = l16.
  const int r0 = rowBase + mrow * 32 + lhalf * 8;
#pragma unroll
  for (int t = 0; t < 2; ++t) {
#pragma unroll
    for (int nt = 0; nt < 8; ++nt) {
      const int oc = oBase + ncol * 128 + nt * 16 + l16;
#pragma unroll
      for (int j = 0; j < 8; ++j)
        out[(size_t)(r0 + t * 16 + j) * OUTPUT_DIM + oc] = acc[t * 8 + nt][j];
    }
  }
}

// ---------------------------------------------------------------------------
extern "C" void kernel_launch(void* const* d_in, const int* in_sizes, int n_in,
                              void* d_out, int out_size, void* d_ws, size_t ws_size,
                              hipStream_t stream) {
  const float* x      = (const float*)d_in[0];
  const float* coeffs = (const float*)d_in[1];
  float*       out    = (float*)d_out;
  _Float16*    bws    = (_Float16*)d_ws;  // 9*16*512*32*2B = 4.5 MB

  const int n_rows = in_sizes[0] / INPUT_DIM;  // 32768

  const int packTotal = NDEG * NIB * OUTPUT_DIM * KI;
  hipLaunchKernelGGL(hahn_pack_coeffs, dim3((packTotal + 255) / 256), dim3(256),
                     0, stream, coeffs, bws);

  dim3 grid(n_rows / ROWS, OUTPUT_DIM / COLS);  // 256 x 2
  hipLaunchKernelGGL(hahn_wmma, grid, dim3(THREADS), 0, stream, x, bws, out);
}
